// PIC_56203942035623
// MI455X (gfx1250) — matrix operand order repaired and checked by reading, compiled, not run on hardware
//
#include <hip/hip_runtime.h>
#include <hip/hip_bf16.h>
#include <math.h>

typedef __attribute__((ext_vector_type(16))) __bf16 v16bf;
typedef __attribute__((ext_vector_type(8)))  __bf16 v8bf;
typedef __attribute__((ext_vector_type(8)))  float  v8f;
typedef __attribute__((ext_vector_type(4)))  int    v4i;

#define B_    16
#define CIN_  256
#define H_    64
#define W_    64
#define HID_  256
#define OUT_  256
#define HP    66          // H + 2 (padded)
#define WP    66          // W + 2 (padded)
#define CK    288         // channels per tap: 256 data + 1 bias + 31 zero-pad (9 K-chunks of 32)
#define KTOT  (9 * CK)    // 2592
#define LDSP  296         // LDS pitch per position (halves); 592B = 148 dwords -> conflict-free b128

static constexpr size_t XPAD_ELEMS = (size_t)B_ * HP * WP * CK;  // 20,072,448 bf16
static constexpr size_t XPAD_BYTES = XPAD_ELEMS * 2;             // 40,144,896

// ---------------------------------------------------------------------------
// CDNA5 async global->LDS copy (16B per lane) + ASYNCcnt wait.
// Builtin signature (from hipcc diagnostic): param1 = addrspace(1) v4i*,
// so: (global src, lds dst, imm offset, imm cpol).
// ---------------------------------------------------------------------------
typedef __attribute__((address_space(1))) v4i* gp_v4i;
typedef __attribute__((address_space(3))) v4i* lp_v4i;

__device__ __forceinline__ void async_copy16(const __bf16* g, __bf16* l) {
#if __has_builtin(__builtin_amdgcn_global_load_async_to_lds_b128)
    __builtin_amdgcn_global_load_async_to_lds_b128((gp_v4i)g, (lp_v4i)l, 0, 0);
#else
    unsigned lofs = (unsigned)(uintptr_t)l;   // low 32 bits of generic LDS ptr = LDS offset
    asm volatile("global_load_async_to_lds_b128 %0, %1, off"
                 :: "v"(lofs), "v"(g) : "memory");
#endif
}

__device__ __forceinline__ void wait_async0() {
#if __has_builtin(__builtin_amdgcn_s_wait_asynccnt)
    __builtin_amdgcn_s_wait_asynccnt(0);
#else
    asm volatile("s_wait_asynccnt 0x0" ::: "memory");
#endif
}

// ---------------------------------------------------------------------------
// Zero the padded/augmented input buffer (borders must be zero every launch).
// ---------------------------------------------------------------------------
__global__ void PIC_zero_xpad(uint4* __restrict__ p, size_t n16) {
    size_t i = (size_t)blockIdx.x * blockDim.x + threadIdx.x;
    if (i < n16) {
        uint4 z; z.x = 0u; z.y = 0u; z.z = 0u; z.w = 0u;
        p[i] = z;
    }
}

// ---------------------------------------------------------------------------
// NCHW fp32 -> padded NHWC bf16 with appended constant-1 bias channel.
// ---------------------------------------------------------------------------
__global__ __launch_bounds__(288) void PIC_pad_convert(const float* __restrict__ x,
                                                       __bf16* __restrict__ xpad) {
    __shared__ __bf16 tile[CIN_][W_ + 2];
    const int bh = blockIdx.x;
    const int b  = bh >> 6;
    const int h  = bh & 63;
    const int tid = threadIdx.x;

    for (int i = tid; i < CIN_ * W_; i += 288) {
        const int c = i >> 6;
        const int w = i & 63;
        tile[c][w] = (__bf16)x[(((size_t)b * CIN_ + c) * H_ + h) * W_ + w];
    }
    __syncthreads();

    const int c = tid;
    __bf16* dst = xpad + (((size_t)b * HP + (h + 1)) * WP + 1) * CK + c;
    for (int w = 0; w < W_; ++w) {
        __bf16 v;
        if (c < CIN_)        v = tile[c][w];
        else if (c == CIN_)  v = (__bf16)1.0f;   // bias channel
        else                 v = (__bf16)0.0f;   // K padding
        dst[(size_t)w * CK] = v;
    }
}

// ---------------------------------------------------------------------------
// Fuse Wf @ W9 into WeffT[o][k*288 + c] (bf16), bias folded into c==256 row.
// ---------------------------------------------------------------------------
__global__ __launch_bounds__(288) void PIC_fuse_weights(const float* __restrict__ W9,
                                                        const float* __restrict__ b9,
                                                        const float* __restrict__ Wf,
                                                        __bf16* __restrict__ weffT) {
    const int o = blockIdx.x / 9;
    const int k = blockIdx.x % 9;
    const int c = threadIdx.x;
    const float* wfrow = Wf + (size_t)o * (9 * HID_) + (size_t)k * HID_;
    float acc = 0.f;
    if (c < CIN_) {
        const float* w9col = W9 + (size_t)k * HID_ * CIN_ + c;
        #pragma unroll 4
        for (int hid = 0; hid < HID_; ++hid)
            acc += wfrow[hid] * w9col[(size_t)hid * CIN_];
    } else if (c == CIN_) {
        const float* b9k = b9 + (size_t)k * HID_;
        #pragma unroll 4
        for (int hid = 0; hid < HID_; ++hid)
            acc += wfrow[hid] * b9k[hid];
    }
    weffT[(size_t)o * KTOT + (size_t)k * CK + c] = (__bf16)acc;
}

// ---------------------------------------------------------------------------
// Main implicit-GEMM 3x3 conv via WMMA bf16 -> f32.
// Workgroup = one (b,h) output row: 256 outs x 64 pos, 8 waves of 2x4 tiles.
// Per t in {0,1,2}: async-stage padded input row (h+2-t) [66 pos x 288 ch]
// into LDS once, then 3 taps x 9 K-chunks of wmma against it.
//  A frag (global, L2-resident): lane l16 = row o; 2x16B runs at K +0/+16 (+8 hi half).
//  B frag (LDS): lane l16 = column pos; 2x16B runs at K +0/+16 (+8 hi half).
// ---------------------------------------------------------------------------
__global__ __launch_bounds__(256) void PIC_conv3x3_wmma(const __bf16* __restrict__ xpad,
                                                        const __bf16* __restrict__ weffT,
                                                        const float* __restrict__ bfv,
                                                        float* __restrict__ out) {
    __shared__ __bf16 ldsrow[WP * LDSP];     // 66 * 296 * 2B = 39,072 B

    const int tid  = threadIdx.x;
    const int lane = tid & 31;
    const int wv   = tid >> 5;
    const int bh   = blockIdx.x;
    const int b    = bh >> 6;
    const int h    = bh & 63;
    const int obase = wv * 32;
    const int l16  = lane & 15;
    const int lh   = lane >> 4;

    v8f acc[2][4] = {};

    for (int t = 0; t < 3; ++t) {
        __syncthreads();   // previous row fully consumed before overwrite
        // Stage padded input row (h + 2 - t): 66 pos x 288 ch = 2376 x 16B pieces.
        const __bf16* grow = xpad + (((size_t)b * HP + (h + 2 - t)) * WP) * CK;
        for (int p = tid; p < WP * 36; p += 256) {
            const int pos = p / 36;
            const int sub = p % 36;
            async_copy16(grow + (size_t)pos * CK + sub * 8,
                         ldsrow + (size_t)pos * LDSP + sub * 8);
        }
        wait_async0();
        __syncthreads();

        for (int l = 0; l < 3; ++l) {
            const int k = t * 3 + l;
            const __bf16* a0p = weffT + (size_t)(obase + l16) * KTOT + (size_t)k * CK + lh * 8;
            const __bf16* a1p = a0p + (size_t)16 * KTOT;
            const __bf16* bp  = ldsrow + (size_t)(l16 + 2 - l) * LDSP + lh * 8;

            for (int c0 = 0; c0 < CK; c0 += 32) {
                v8bf a0lo = *(const v8bf*)(a0p + c0);
                v8bf a0hi = *(const v8bf*)(a0p + c0 + 16);
                v8bf a1lo = *(const v8bf*)(a1p + c0);
                v8bf a1hi = *(const v8bf*)(a1p + c0 + 16);
                v16bf a0 = __builtin_shufflevector(a0lo, a0hi, 0,1,2,3,4,5,6,7,8,9,10,11,12,13,14,15);
                v16bf a1 = __builtin_shufflevector(a1lo, a1hi, 0,1,2,3,4,5,6,7,8,9,10,11,12,13,14,15);

                v16bf bt[4];
                #pragma unroll
                for (int nt = 0; nt < 4; ++nt) {
                    const __bf16* bpn = bp + (size_t)(nt * 16) * LDSP + c0;
                    v8bf blo = *(const v8bf*)(bpn);
                    v8bf bhi = *(const v8bf*)(bpn + 16);
                    bt[nt] = __builtin_shufflevector(blo, bhi, 0,1,2,3,4,5,6,7,8,9,10,11,12,13,14,15);
                }

                acc[0][0] = __builtin_amdgcn_wmma_f32_16x16x32_bf16(false, a0, false, bt[0], (short)0, acc[0][0], false, false);
                acc[0][1] = __builtin_amdgcn_wmma_f32_16x16x32_bf16(false, a0, false, bt[1], (short)0, acc[0][1], false, false);
                acc[0][2] = __builtin_amdgcn_wmma_f32_16x16x32_bf16(false, a0, false, bt[2], (short)0, acc[0][2], false, false);
                acc[0][3] = __builtin_amdgcn_wmma_f32_16x16x32_bf16(false, a0, false, bt[3], (short)0, acc[0][3], false, false);
                acc[1][0] = __builtin_amdgcn_wmma_f32_16x16x32_bf16(false, a1, false, bt[0], (short)0, acc[1][0], false, false);
                acc[1][1] = __builtin_amdgcn_wmma_f32_16x16x32_bf16(false, a1, false, bt[1], (short)0, acc[1][1], false, false);
                acc[1][2] = __builtin_amdgcn_wmma_f32_16x16x32_bf16(false, a1, false, bt[2], (short)0, acc[1][2], false, false);
                acc[1][3] = __builtin_amdgcn_wmma_f32_16x16x32_bf16(false, a1, false, bt[3], (short)0, acc[1][3], false, false);
            }
        }
    }

    // Epilogue: D VGPR r holds row M = r (lanes 0-15) / r+8 (lanes 16-31), N = l16.
    #pragma unroll
    for (int mi = 0; mi < 2; ++mi) {
        #pragma unroll
        for (int r = 0; r < 8; ++r) {
            const int o = obase + mi * 16 + r + lh * 8;
            const float bias = bfv[o];
            float* orow = out + (((size_t)b * OUT_ + o) * H_ + h) * W_;
            #pragma unroll
            for (int nt = 0; nt < 4; ++nt) {
                const float v = acc[mi][nt][r] + bias;
                orow[nt * 16 + l16] = 1.0f / (1.0f + __expf(-v));
            }
        }
    }
}

// ---------------------------------------------------------------------------
extern "C" void kernel_launch(void* const* d_in, const int* in_sizes, int n_in,
                              void* d_out, int out_size, void* d_ws, size_t ws_size,
                              hipStream_t stream) {
    const float* x   = (const float*)d_in[0];   // [B, CIN, H, W]
    const float* W9  = (const float*)d_in[1];   // [9, HID, CIN]
    const float* b9  = (const float*)d_in[2];   // [9, HID]
    const float* Wf  = (const float*)d_in[3];   // [OUT, 9*HID]
    const float* bfv = (const float*)d_in[4];   // [OUT]
    float* out = (float*)d_out;                 // [B, OUT, H, W]

    char* ws = (char*)d_ws;
    __bf16* xpad  = (__bf16*)ws;
    __bf16* weffT = (__bf16*)(ws + ((XPAD_BYTES + 255) & ~(size_t)255));

    const size_t n16 = XPAD_BYTES / 16;
    PIC_zero_xpad<<<(unsigned)((n16 + 255) / 256), 256, 0, stream>>>((uint4*)xpad, n16);
    PIC_pad_convert<<<B_ * H_, 288, 0, stream>>>(x, xpad);
    PIC_fuse_weights<<<OUT_ * 9, 288, 0, stream>>>(W9, b9, Wf, weffT);
    PIC_conv3x3_wmma<<<B_ * H_, 256, 0, stream>>>(xpad, weffT, bfv, out);
}
